// DeepseekV3MoE_13262859010622
// MI455X (gfx1250) — matrix operand order repaired and checked by reading, not compiled, not run on hardware
//
#include <hip/hip_runtime.h>
#include <hip/hip_bf16.h>
#include <math.h>

// ---------------- problem constants ----------------
#define NB      2
#define SS      2048
#define NTOK    (NB * SS)      // 4096 tokens
#define HH      1024
#define II      256
#define EE      256
#define KK      8
#define GG      8
#define TGTOP   4
#define NSH     2
#define CAP     2048           // per-expert token capacity (expected ~128)

// ---------------- WMMA types ----------------
typedef __bf16 v16bf __attribute__((ext_vector_type(16)));
typedef float  v8f   __attribute__((ext_vector_type(8)));

__device__ __forceinline__ v8f wmma_bf16(v16bf a, v16bf b, v8f c) {
    // D = A(16x32 bf16) * B(32x16 bf16) + C(16x16 f32)
    return __builtin_amdgcn_wmma_f32_16x16x32_bf16(
        /*neg_a=*/false, a, /*neg_b=*/false, b,
        /*c_mod=*/(short)0, c, /*reuse_a=*/false, /*reuse_b=*/false);
}

__device__ __forceinline__ float siluf(float v) {
    return v / (1.0f + __expf(-v));
}

// async copy of 16 bytes global -> LDS (no VGPR round trip, ASYNCcnt tracked)
__device__ __forceinline__ void async_copy_b128(unsigned lds_byte_off, const void* gptr) {
    asm volatile("global_load_async_to_lds_b128 %0, %1, off"
                 :: "v"(lds_byte_off), "v"(gptr) : "memory");
}
__device__ __forceinline__ void async_wait0() {
    asm volatile("s_wait_asynccnt 0x0" ::: "memory");
}

// A-fragment (16x32 bf16) from LDS row-major tile [16 x ld]
// layout (ISA 7.12.2): lane r (r=lane&15) holds row M=r; lane-half selects K sub-ranges
__device__ __forceinline__ v16bf load_a_lds(const __bf16* lds, int ld, int k0, int lane) {
    int half = lane >> 4, r = lane & 15;
    v16bf a;
#pragma unroll
    for (int i = 0; i < 16; ++i) {
        int k = k0 + half * 16 + i;
        a[i] = lds[r * ld + k];
    }
    return a;
}

// B-fragment (32x16 bf16) from global fp32 row-major W[K x ldw], converting to bf16.
// Weights are streamed once -> non-temporal loads (TH_LOAD_NT) to keep L2 for activations.
// lanes 0-15 hold K=k0..k0+15 for column N=n0+r; lanes 16-31 hold K=k0+16..k0+31.
__device__ __forceinline__ v16bf load_b_global(const float* W, int ldw, int k0, int n0, int lane) {
    int half = lane >> 4, r = lane & 15;
    v16bf b;
#pragma unroll
    for (int i = 0; i < 16; ++i) {
        int k = k0 + half * 16 + i;
        b[i] = (__bf16)__builtin_nontemporal_load(&W[(size_t)k * ldw + (n0 + r)]);
    }
    return b;
}

// ---------------- kernel 0: zero expert counters ----------------
__global__ void init_counts_kernel(int* counts) {
    int i = threadIdx.x;
    if (i < EE) counts[i] = 0;
}

// ---------------- kernel 1: fp32 -> bf16 activation pre-convert ----------------
__global__ void __launch_bounds__(256) convert_x_kernel(const float* __restrict__ x,
                                                        __bf16* __restrict__ xbf) {
    int i = (blockIdx.x * 256 + threadIdx.x) * 4;   // N*H = 4194304, grid covers exactly
    float4 v = *(const float4*)(x + i);
    xbf[i + 0] = (__bf16)v.x;
    xbf[i + 1] = (__bf16)v.y;
    xbf[i + 2] = (__bf16)v.z;
    xbf[i + 3] = (__bf16)v.w;
}

// ---------------- kernel 2: router (logits + grouped top-k + scatter) ----------------
__global__ void __launch_bounds__(256) router_kernel(const float* __restrict__ x,
                                                     const float* __restrict__ Wg,
                                                     int* __restrict__ counts,
                                                     int* __restrict__ etok,
                                                     float* __restrict__ ewt) {
    __shared__ float sl[EE];
    __shared__ float cl[GG * TGTOP];
    __shared__ int   ci[GG * TGTOP];
    __shared__ int   sIdx[KK];
    __shared__ float sWt[KK];

    int t = blockIdx.x;
    int e = threadIdx.x;

    // logit[t][e] = dot(x[t], Wg[e])
    const float4* xr = (const float4*)(x + (size_t)t * HH);
    const float4* wr = (const float4*)(Wg + (size_t)e * HH);
    float acc = 0.0f;
#pragma unroll 8
    for (int i = 0; i < HH / 4; ++i) {
        float4 a = xr[i], b = wr[i];
        acc += a.x * b.x + a.y * b.y + a.z * b.z + a.w * b.w;
    }
    sl[e] = acc;
    __syncthreads();

    // per-group top-4 (8 groups of 32)
    if (threadIdx.x < GG) {
        int g = threadIdx.x;
        float loc[32];
#pragma unroll
        for (int i = 0; i < 32; ++i) loc[i] = sl[g * 32 + i];
        for (int j = 0; j < TGTOP; ++j) {
            float best = -3.0e38f; int bi = 0;
            for (int i = 0; i < 32; ++i)
                if (loc[i] > best) { best = loc[i]; bi = i; }
            cl[g * TGTOP + j] = best;
            ci[g * TGTOP + j] = g * 32 + bi;
            loc[bi] = -3.0e38f;
        }
    }
    __syncthreads();

    // global top-8 over 32 candidates, then softmax
    if (threadIdx.x == 0) {
        float tl[KK]; int ti[KK];
        float tmp[GG * TGTOP];
        for (int i = 0; i < GG * TGTOP; ++i) tmp[i] = cl[i];
        for (int j = 0; j < KK; ++j) {
            float best = -3.0e38f; int bi = 0;
            for (int i = 0; i < GG * TGTOP; ++i)
                if (tmp[i] > best) { best = tmp[i]; bi = i; }
            tl[j] = best; ti[j] = ci[bi];
            tmp[bi] = -3.0e38f;
        }
        float mx = tl[0], sum = 0.0f;
        float w[KK];
        for (int j = 0; j < KK; ++j) { w[j] = __expf(tl[j] - mx); sum += w[j]; }
        float inv = 1.0f / sum;
        for (int j = 0; j < KK; ++j) { sIdx[j] = ti[j]; sWt[j] = w[j] * inv; }
    }
    __syncthreads();

    // scatter into per-expert lists
    if (threadIdx.x < KK) {
        int j = threadIdx.x;
        int ex = sIdx[j];
        int pos = atomicAdd(&counts[ex], 1);
        if (pos < CAP) {
            etok[ex * CAP + pos] = t;
            ewt[ex * CAP + pos] = sWt[j];
        }
    }
}

// ---------------- kernel 3: shared experts (dense fused MLP, writes d_out) ----------------
__global__ void __launch_bounds__(256) shared_mlp_kernel(const __bf16* __restrict__ xbf,
                                                         const float* __restrict__ Wsgu,
                                                         const float* __restrict__ Wsd,
                                                         float* __restrict__ out) {
    __shared__ __bf16 sX[16 * HH];   // 32 KB
    __shared__ __bf16 sH[16 * II];   // 8 KB

    int t0 = blockIdx.x * 16;
    int tid = threadIdx.x, wave = tid >> 5, lane = tid & 31;
    int half = lane >> 4, r = lane & 15;

    // stage 16x1024 bf16 x-tile into LDS via async DMA (2048 b128, 8 per thread)
    {
        const uint4* src = (const uint4*)(xbf + (size_t)t0 * HH);
        unsigned sx_base = (unsigned)(uintptr_t)(void*)&sX[0];
#pragma unroll
        for (int i = 0; i < 8; ++i) {
            int idx = tid + 256 * i;
            async_copy_b128(sx_base + idx * 16, (const void*)(src + idx));
        }
        async_wait0();
    }
    __syncthreads();

    v8f dacc[8];
#pragma unroll
    for (int f = 0; f < 8; ++f) dacc[f] = (v8f){};

    for (int s = 0; s < NSH; ++s) {
        const float* Wgu = Wsgu + (size_t)s * HH * (2 * II);
        const float* Wd  = Wsd  + (size_t)s * II * HH;

        // ---- GEMM 1: gu = x @ Wgu ; wave owns 32 gate cols + matching 32 up cols ----
        int ng = wave * 32;
        v8f g0 = (v8f){}, g1 = (v8f){}, u0 = (v8f){}, u1 = (v8f){};
        for (int k0 = 0; k0 < HH; k0 += 32) {
            if (k0 + 32 < HH)
                __builtin_prefetch(&Wgu[(size_t)(k0 + 32) * (2 * II) + ng], 0, 1);
            v16bf a  = load_a_lds(sX, HH, k0, lane);
            v16bf b0 = load_b_global(Wgu, 2 * II, k0, ng, lane);
            v16bf b1 = load_b_global(Wgu, 2 * II, k0, ng + 16, lane);
            v16bf b2 = load_b_global(Wgu, 2 * II, k0, II + ng, lane);
            v16bf b3 = load_b_global(Wgu, 2 * II, k0, II + ng + 16, lane);
            g0 = wmma_bf16(a, b0, g0);
            g1 = wmma_bf16(a, b1, g1);
            u0 = wmma_bf16(a, b2, u0);
            u1 = wmma_bf16(a, b3, u1);
        }

        // ---- silu(g)*u -> sH (bf16, 16 x 256) ----
#pragma unroll
        for (int i = 0; i < 8; ++i) {
            int m = i + half * 8;
            sH[m * II + ng + r]      = (__bf16)(siluf(g0[i]) * u0[i]);
            sH[m * II + ng + 16 + r] = (__bf16)(siluf(g1[i]) * u1[i]);
        }
        __syncthreads();

        // ---- GEMM 2: y += hmid @ Wd ; wave owns 128 output cols ----
        int nd = wave * 128;
        for (int k0 = 0; k0 < II; k0 += 32) {
            v16bf a = load_a_lds(sH, II, k0, lane);
#pragma unroll
            for (int f = 0; f < 8; ++f) {
                v16bf b = load_b_global(Wd, HH, k0, nd + f * 16, lane);
                dacc[f] = wmma_bf16(a, b, dacc[f]);
            }
        }
        __syncthreads();   // sH reused next shared expert
    }

    // write (shared1 + shared2)/NS
    int nd = wave * 128;
#pragma unroll
    for (int f = 0; f < 8; ++f)
#pragma unroll
        for (int i = 0; i < 8; ++i) {
            int m = i + half * 8;
            out[(size_t)(t0 + m) * HH + nd + f * 16 + r] = dacc[f][i] * (1.0f / NSH);
        }
}

// ---------------- kernel 4: routed experts (expert-stationary, atomic add) ----------------
__global__ void __launch_bounds__(256) routed_mlp_kernel(const __bf16* __restrict__ xbf,
                                                         const float* __restrict__ Wgu_all,
                                                         const float* __restrict__ Wd_all,
                                                         const int* __restrict__ counts,
                                                         const int* __restrict__ etok,
                                                         const float* __restrict__ ewt,
                                                         float* __restrict__ out) {
    __shared__ __bf16 sX[16 * HH];   // gathered token tile
    __shared__ __bf16 sH[16 * II];
    __shared__ int   sTok[16];
    __shared__ float sW[16];

    int e = blockIdx.x;
    int cnt = counts[e];
    if (cnt > CAP) cnt = CAP;
    if (cnt == 0) return;

    const float* Wgu = Wgu_all + (size_t)e * HH * (2 * II);
    const float* Wd  = Wd_all  + (size_t)e * II * HH;

    int tid = threadIdx.x, wave = tid >> 5, lane = tid & 31;
    int half = lane >> 4, r = lane & 15;
    unsigned sx_base = (unsigned)(uintptr_t)(void*)&sX[0];

    for (int tile0 = 0; tile0 < cnt; tile0 += 16) {
        // ---- gather 16 token rows via async DMA (16 threads per row, 8 b128 each) ----
        {
            int row = tid >> 4, sub = tid & 15;
            int idx = tile0 + row;
            int tok = (idx < cnt) ? etok[e * CAP + idx] : -1;
            if (sub == 0) {
                sTok[row] = tok;
                sW[row] = (idx < cnt) ? ewt[e * CAP + idx] : 0.0f;
            }
            if (tok >= 0) {
                const uint4* src = (const uint4*)(xbf + (size_t)tok * HH);
#pragma unroll
                for (int i = 0; i < 8; ++i) {
                    int idx4 = row * 128 + sub + 16 * i;          // uint4 index in sX
                    async_copy_b128(sx_base + idx4 * 16, (const void*)(src + sub + 16 * i));
                }
            } else {
                uint4 z = {0, 0, 0, 0};
                uint4* dst = (uint4*)(sX + row * HH);
#pragma unroll
                for (int i = 0; i < 8; ++i) dst[sub + 16 * i] = z;
            }
            async_wait0();
        }
        __syncthreads();

        // ---- GEMM 1: gate/up ----
        int ng = wave * 32;
        v8f g0 = (v8f){}, g1 = (v8f){}, u0 = (v8f){}, u1 = (v8f){};
        for (int k0 = 0; k0 < HH; k0 += 32) {
            if (k0 + 32 < HH)
                __builtin_prefetch(&Wgu[(size_t)(k0 + 32) * (2 * II) + ng], 0, 1);
            v16bf a  = load_a_lds(sX, HH, k0, lane);
            v16bf b0 = load_b_global(Wgu, 2 * II, k0, ng, lane);
            v16bf b1 = load_b_global(Wgu, 2 * II, k0, ng + 16, lane);
            v16bf b2 = load_b_global(Wgu, 2 * II, k0, II + ng, lane);
            v16bf b3 = load_b_global(Wgu, 2 * II, k0, II + ng + 16, lane);
            g0 = wmma_bf16(a, b0, g0);
            g1 = wmma_bf16(a, b1, g1);
            u0 = wmma_bf16(a, b2, u0);
            u1 = wmma_bf16(a, b3, u1);
        }

#pragma unroll
        for (int i = 0; i < 8; ++i) {
            int m = i + half * 8;
            sH[m * II + ng + r]      = (__bf16)(siluf(g0[i]) * u0[i]);
            sH[m * II + ng + 16 + r] = (__bf16)(siluf(g1[i]) * u1[i]);
        }
        __syncthreads();

        // ---- GEMM 2: down ----
        int nd = wave * 128;
        v8f dacc[8];
#pragma unroll
        for (int f = 0; f < 8; ++f) dacc[f] = (v8f){};
        for (int k0 = 0; k0 < II; k0 += 32) {
            v16bf a = load_a_lds(sH, II, k0, lane);
#pragma unroll
            for (int f = 0; f < 8; ++f) {
                v16bf b = load_b_global(Wd, HH, k0, nd + f * 16, lane);
                dacc[f] = wmma_bf16(a, b, dacc[f]);
            }
        }

        // ---- weighted atomic scatter-add into output ----
#pragma unroll
        for (int i = 0; i < 8; ++i) {
            int m = i + half * 8;
            int tok = sTok[m];
            if (tok >= 0) {
                float w = sW[m];
#pragma unroll
                for (int f = 0; f < 8; ++f)
                    atomicAdd(&out[(size_t)tok * HH + nd + f * 16 + r], dacc[f][i] * w);
            }
        }
        __syncthreads();   // before next tile overwrites sX/sH/sTok
    }
}

// ---------------- host launcher ----------------
extern "C" void kernel_launch(void* const* d_in, const int* in_sizes, int n_in,
                              void* d_out, int out_size, void* d_ws, size_t ws_size,
                              hipStream_t stream) {
    const float* x     = (const float*)d_in[0];   // [B,S,H]
    const float* Wg    = (const float*)d_in[1];   // [E,H]
    const float* Wgu   = (const float*)d_in[2];   // [E,H,2I]
    const float* Wd    = (const float*)d_in[3];   // [E,I,H]
    const float* Wsgu  = (const float*)d_in[4];   // [NS,H,2I]
    const float* Wsd   = (const float*)d_in[5];   // [NS,I,H]
    float* out = (float*)d_out;

    // workspace layout
    int*    counts = (int*)d_ws;                        // E
    int*    etok   = counts + EE;                       // E*CAP
    float*  ewt    = (float*)(etok + EE * CAP);         // E*CAP
    __bf16* xbf    = (__bf16*)(ewt + EE * CAP);         // N*H bf16

    init_counts_kernel<<<1, 256, 0, stream>>>(counts);
    convert_x_kernel<<<(NTOK * HH) / 1024, 256, 0, stream>>>(x, xbf);
    router_kernel<<<NTOK, 256, 0, stream>>>(x, Wg, counts, etok, ewt);
    shared_mlp_kernel<<<NTOK / 16, 256, 0, stream>>>(xbf, Wsgu, Wsd, out);
    routed_mlp_kernel<<<EE, 256, 0, stream>>>(xbf, Wgu, Wd, counts, etok, ewt, out);
}
